// ComplexDotLayer_89610197663823
// MI455X (gfx1250) — compile-verified, matrix operand order checked by reading, not run on hardware
//
#include <hip/hip_runtime.h>
#include <hip/hip_bf16.h>
#include <stdint.h>

typedef __attribute__((ext_vector_type(16))) __bf16 v16bf;
typedef __attribute__((ext_vector_type(8)))  float  v8f;

#define CIN   128
#define COUT  256
#define BB    64
#define NPIX  1024
#define WSTRIDE 136            // ushorts per LDS row: 128 + 8 pad (272B, 16B-aligned)

// LDS layout (bytes); all offsets multiples of 16
#define OFF_WTR   0
#define OFF_WTI   (OFF_WTR + COUT*WSTRIDE*2)     // 69632
#define OFF_XR    (OFF_WTI + COUT*WSTRIDE*2)     // 139264
#define OFF_XI    (OFF_XR  + BB*WSTRIDE*2)       // 156672
#define OFF_XIN   (OFF_XI  + BB*WSTRIDE*2)       // 174080
#define OFF_STATS (OFF_XIN + BB*WSTRIDE*2)       // 191488
#define LDS_BYTES (OFF_STATS + 1024*4)           // 195584 < 320KB

static __device__ __forceinline__ unsigned short f2bf(float f) {
  union { float f; unsigned int u; } c; c.f = f;
  unsigned int r = (c.u + 0x7FFFu + ((c.u >> 16) & 1u)) >> 16;  // round-to-nearest-even
  return (unsigned short)r;
}

union Frag { v16bf v; uint4 q[2]; };

__global__ __launch_bounds__(256) void zero_stats_kernel(float* s) {
  s[blockIdx.x * 256 + threadIdx.x] = 0.0f;
}

__global__ __launch_bounds__(256) void complex_conv_wmma_kernel(
    const float* __restrict__ xr_g, const float* __restrict__ xi_g,
    const float* __restrict__ wt_g,
    float* __restrict__ outR, float* __restrict__ outI,
    float* __restrict__ stats_g) {
  extern __shared__ char smem[];
  unsigned short* wtr = (unsigned short*)(smem + OFF_WTR);   // [COUT][WSTRIDE]  wR^T bf16
  unsigned short* wti = (unsigned short*)(smem + OFF_WTI);   // [COUT][WSTRIDE]  wI^T bf16
  unsigned short* xr  = (unsigned short*)(smem + OFF_XR);    // [BB][WSTRIDE]
  unsigned short* xi  = (unsigned short*)(smem + OFF_XI);
  unsigned short* xin = (unsigned short*)(smem + OFF_XIN);   // -x_imag
  float* stats = (float*)(smem + OFF_STATS);                 // sumR,sqR,sumI,sqI [4][256]

  const int t   = threadIdx.x;
  const int pix = blockIdx.x;          // h*32 + w
  const int h   = pix >> 5;
  const int w   = pix & 31;

  // zero per-WG stats
  #pragma unroll
  for (int k = 0; k < 4; ++k) stats[t + 256 * k] = 0.0f;

  // ---- Phase 1: analytic DFT of the 3x3 filter at frequency (h,w), bf16 into LDS ----
  float cs[9], sn[9];
  #pragma unroll
  for (int ij = 0; ij < 9; ++ij) {
    const int i = ij / 3, j = ij % 3;
    const float th = (float)(h * i + w * j) * 0.19634954084936207f;  // 2*pi/32
    cs[ij] = __cosf(th);
    sn[ij] = __sinf(th);
  }
  {
    const int o = t;                   // coalesced over Cout (innermost dim of wt)
    for (int c = 0; c < CIN; ++c) {
      float aR = 0.0f, aI = 0.0f;
      #pragma unroll
      for (int ij = 0; ij < 9; ++ij) {
        const float wv = wt_g[(ij * CIN + c) * COUT + o];
        aR += cs[ij] * wv;             // Re(e^{-i th}) =  cos
        aI -= sn[ij] * wv;             // Im(e^{-i th}) = -sin
      }
      wtr[o * WSTRIDE + c] = f2bf(aR);
      wti[o * WSTRIDE + c] = f2bf(aI);
    }
  }

  // ---- Phase 2: stage x tiles (B=64, Cin=128) as bf16 ----
  {
    const float4* xr4 = (const float4*)xr_g;
    const float4* xi4 = (const float4*)xi_g;
    #pragma unroll
    for (int k = 0; k < 8; ++k) {
      const int e  = t + 256 * k;      // float4 index in tile, 0..2047
      const int b  = e >> 5;
      const int c4 = e & 31;
      const int gi = b * 32768 + pix * 32 + c4;   // (b*131072 + pix*128 + c)/4
      const float4 vr = xr4[gi];
      const float4 vi = xi4[gi];
      uint2 pr, pi, pn;
      pr.x = (unsigned)f2bf(vr.x) | ((unsigned)f2bf(vr.y) << 16);
      pr.y = (unsigned)f2bf(vr.z) | ((unsigned)f2bf(vr.w) << 16);
      pi.x = (unsigned)f2bf(vi.x) | ((unsigned)f2bf(vi.y) << 16);
      pi.y = (unsigned)f2bf(vi.z) | ((unsigned)f2bf(vi.w) << 16);
      pn.x = (unsigned)f2bf(-vi.x) | ((unsigned)f2bf(-vi.y) << 16);
      pn.y = (unsigned)f2bf(-vi.z) | ((unsigned)f2bf(-vi.w) << 16);
      const int lo = b * WSTRIDE + c4 * 4;        // 8B-aligned
      *(uint2*)(&xr[lo])  = pr;
      *(uint2*)(&xi[lo])  = pi;
      *(uint2*)(&xin[lo]) = pn;
    }
  }
  __syncthreads();

  // ---- Phase 3: WMMA. wave -> (16 batch rows) x (128 Cout cols) slab ----
  const int lane  = t & 31;
  const int wave  = t >> 5;
  const int mb    = (wave & 3) * 16;   // batch-row base
  const int chalf = wave >> 2;         // Cout half
  const int lrow  = lane & 15;
  const int lh    = lane >> 4;         // 16-bit A/B layout: K = [8h..8h+7, 16+8h..]

  Frag aR[4], aI[4], aN[4];
  #pragma unroll
  for (int kc = 0; kc < 4; ++kc) {
    const int klo = kc * 32 + 8 * lh;
    const int ro  = (mb + lrow) * (WSTRIDE * 2);
    aR[kc].q[0] = *(const uint4*)(smem + OFF_XR  + ro + klo * 2);
    aR[kc].q[1] = *(const uint4*)(smem + OFF_XR  + ro + (klo + 16) * 2);
    aI[kc].q[0] = *(const uint4*)(smem + OFF_XI  + ro + klo * 2);
    aI[kc].q[1] = *(const uint4*)(smem + OFF_XI  + ro + (klo + 16) * 2);
    aN[kc].q[0] = *(const uint4*)(smem + OFF_XIN + ro + klo * 2);
    aN[kc].q[1] = *(const uint4*)(smem + OFF_XIN + ro + (klo + 16) * 2);
  }

  for (int cb = 0; cb < 8; ++cb) {
    const int cbase = chalf * 128 + cb * 16;
    const int orow  = cbase + lrow;
    Frag bR[4], bI[4];
    #pragma unroll
    for (int kc = 0; kc < 4; ++kc) {
      const int klo = kc * 32 + 8 * lh;
      const int ro  = orow * (WSTRIDE * 2);
      bR[kc].q[0] = *(const uint4*)(smem + OFF_WTR + ro + klo * 2);
      bR[kc].q[1] = *(const uint4*)(smem + OFF_WTR + ro + (klo + 16) * 2);
      bI[kc].q[0] = *(const uint4*)(smem + OFF_WTI + ro + klo * 2);
      bI[kc].q[1] = *(const uint4*)(smem + OFF_WTI + ro + (klo + 16) * 2);
    }
    v8f accR = {};
    v8f accI = {};
    #pragma unroll
    for (int kc = 0; kc < 4; ++kc) {
      // outR = xR*wR + xI*wI ; outI = xR*wI + (-xI)*wR
      accR = __builtin_amdgcn_wmma_f32_16x16x32_bf16(false, aR[kc].v, false, bR[kc].v, (short)0, accR, false, false);
      accR = __builtin_amdgcn_wmma_f32_16x16x32_bf16(false, aI[kc].v, false, bI[kc].v, (short)0, accR, false, false);
      accI = __builtin_amdgcn_wmma_f32_16x16x32_bf16(false, aR[kc].v, false, bI[kc].v, (short)0, accI, false, false);
      accI = __builtin_amdgcn_wmma_f32_16x16x32_bf16(false, aN[kc].v, false, bR[kc].v, (short)0, accI, false, false);
    }
    // store (lanes span consecutive Cout -> coalesced) + per-channel stats
    const int n = cbase + lrow;
    float* pR = outR + (size_t)pix * 256 + n;
    float* pI = outI + (size_t)pix * 256 + n;
    const int boff = mb + 8 * lh;      // lanes 0-15: rows v .. ; lanes 16-31: rows v+8
    float sR = 0.f, sR2 = 0.f, sI = 0.f, sI2 = 0.f;
    #pragma unroll
    for (int v = 0; v < 8; ++v) {
      const float vr = accR[v];
      const float vi = accI[v];
      pR[(size_t)(boff + v) * 262144] = vr;   // 262144 = H*W*Cout
      pI[(size_t)(boff + v) * 262144] = vi;
      sR += vr; sR2 += vr * vr;
      sI += vi; sI2 += vi * vi;
    }
    atomicAdd(&stats[n],       sR);
    atomicAdd(&stats[256 + n], sR2);
    atomicAdd(&stats[512 + n], sI);
    atomicAdd(&stats[768 + n], sI2);
  }
  __syncthreads();
  #pragma unroll
  for (int k = 0; k < 4; ++k)
    atomicAdd(&stats_g[t + 256 * k], stats[t + 256 * k]);
}

__global__ __launch_bounds__(256) void finalize_stats_kernel(
    float* __restrict__ ws,
    const float* __restrict__ gr, const float* __restrict__ br,
    const float* __restrict__ gi, const float* __restrict__ bi) {
  const int t = threadIdx.x;
  const float invN = 1.0f / 65536.0f;          // B*H*W
  const float mR = ws[t] * invN;
  const float vR = ws[256 + t] * invN - mR * mR;
  const float sR = gr[t] * rsqrtf(vR + 1e-3f);
  ws[1024 + t] = sR;
  ws[1280 + t] = br[t] - mR * sR;
  const float mI = ws[512 + t] * invN;
  const float vI = ws[768 + t] * invN - mI * mI;
  const float sI = gi[t] * rsqrtf(vI + 1e-3f);
  ws[1536 + t] = sI;
  ws[1792 + t] = bi[t] - mI * sI;
}

__global__ __launch_bounds__(256) void apply_bn_kernel(
    float* __restrict__ out, const float* __restrict__ ws) {
  const size_t i4 = (size_t)blockIdx.x * 256 + threadIdx.x;   // 8388608 float4s
  float4* o4 = (float4*)out;
  float4 v = o4[i4];
  const int half = (i4 >= 4194304u) ? 1 : 0;                  // outR vs outI
  const int o = (int)(i4 & 63) * 4;
  const float4 sc = *(const float4*)(ws + 1024 + 512 * half + o);
  const float4 bi = *(const float4*)(ws + 1280 + 512 * half + o);
  v.x = v.x * sc.x + bi.x;
  v.y = v.y * sc.y + bi.y;
  v.z = v.z * sc.z + bi.z;
  v.w = v.w * sc.w + bi.w;
  o4[i4] = v;
}

extern "C" void kernel_launch(void* const* d_in, const int* in_sizes, int n_in,
                              void* d_out, int out_size, void* d_ws, size_t ws_size,
                              hipStream_t stream) {
  (void)in_sizes; (void)n_in; (void)out_size; (void)ws_size;
  const float* xr = (const float*)d_in[0];
  const float* xi = (const float*)d_in[1];
  const float* wt = (const float*)d_in[2];
  const float* gr = (const float*)d_in[3];
  const float* br = (const float*)d_in[4];
  const float* gi = (const float*)d_in[5];
  const float* bi = (const float*)d_in[6];
  float* outR = (float*)d_out;
  float* outI = outR + 16777216;       // B*H*W*Cout
  float* ws   = (float*)d_ws;

  zero_stats_kernel<<<4, 256, 0, stream>>>(ws);
  complex_conv_wmma_kernel<<<NPIX, 256, LDS_BYTES, stream>>>(xr, xi, wt, outR, outI, ws);
  finalize_stats_kernel<<<1, 256, 0, stream>>>(ws, gr, br, gi, bi);
  apply_bn_kernel<<<32768, 256, 0, stream>>>((float*)d_out, ws);
}
